// GQA_20100446945551
// MI455X (gfx1250) — matrix operand order repaired and checked by reading, not compiled, not run on hardware
//
#include <hip/hip_runtime.h>
#include <stdint.h>

// ---------------------------------------------------------------------------
// GQA block for MI455X (gfx1250, wave32).
//  * one-time fp32->bf16 conversion (+ weight transpose to [N][K])
//  * all GEMMs + attention matmuls via v_wmma_f32_16x16x32_bf16
//  * LDS staging via Tensor Data Mover (tensor_load_to_lds, TENSORcnt),
//    double-buffered for async copy/compute overlap
// ---------------------------------------------------------------------------

#define D_MODEL   2048
#define NUM_HEADS 32
#define NUM_GROUPS 8
#define HEAD_DIM  64
#define HPG       4
#define BATCH     2
#define SEQ       2048
#define KV_DIM    (HEAD_DIM * NUM_GROUPS)   // 512
#define ROWS      (BATCH * SEQ)             // 4096

typedef unsigned int u32;
typedef unsigned long long u64;
typedef __attribute__((ext_vector_type(16))) __bf16 v16bf;
typedef __attribute__((ext_vector_type(8)))  __bf16 v8bf;
typedef __attribute__((ext_vector_type(8)))  float  v8f;
typedef __attribute__((ext_vector_type(4)))  u32    v4u;
typedef __attribute__((ext_vector_type(4)))  int    v4i;
typedef __attribute__((ext_vector_type(8)))  int    v8i;

// --- WMMA fragment helpers (layouts per CDNA5 ISA 7.12.2, wave32) -----------
__device__ __forceinline__ v16bf frag_pair(const __bf16* p0, const __bf16* p1) {
  union { v16bf v; v8bf h[2]; } u;
  u.h[0] = *(const v8bf*)p0;
  u.h[1] = *(const v8bf*)p1;
  return u.v;
}
// A 16x32 [m][k]: lane (half,m) holds K = {8h..8h+7, 16+8h..16+8h+7} of row m
__device__ __forceinline__ v16bf load_frag_a(const __bf16* row, int half) {
  return frag_pair(row + 8 * half, row + 16 + 8 * half);
}
// B 32x16 stored transposed [n][k]: lane holds K = {8h..8h+7, 16+8h..16+8h+7} of col n
__device__ __forceinline__ v16bf load_frag_b(const __bf16* row, int half) {
  return frag_pair(row + 8 * half, row + 16 + 8 * half);
}
__device__ __forceinline__ v8f wmma_bf16(v16bf a, v16bf b, v8f c) {
  return __builtin_amdgcn_wmma_f32_16x16x32_bf16(
      false, a, false, b, (short)0, c, false, false);
}

// --- Tensor Data Mover: 2D bf16 tile -> LDS (D# per ISA 8.3-8.6) ------------
// pad_interval code: pad every (1<<code)*8 bytes; pad_amount code: (code+1) DW.
__device__ __forceinline__ void tdm_load_tile_2d(
    u32 lds_byte_addr, const void* tile_ptr,
    u32 tile_w_elems, u32 tile_h,
    u32 row_stride_elems, u32 tensor_w_elems, u32 tensor_h,
    u32 pad_interval_code, u32 pad_amount_code)
{
  u64 ga = (u64)(uintptr_t)tile_ptr;
  v4u g0;
  g0.x = 1u;                                    // count=1, user descriptor
  g0.y = lds_byte_addr;                         // lds_addr
  g0.z = (u32)ga;                               // global_addr[31:0]
  g0.w = (u32)(ga >> 32) | (2u << 30);          // global_addr[56:32] | type=2
  v8i g1;
  g1[0] = (int)((1u << 16)                      // data_size = 1 -> 2 bytes
              | (1u << 20)                      // pad_enable
              | (pad_interval_code << 22)
              | (pad_amount_code << 25));
  g1[1] = (int)((tensor_w_elems & 0xFFFFu) << 16);                 // dim0 lo
  g1[2] = (int)((tensor_w_elems >> 16) | ((tensor_h & 0xFFFFu) << 16));
  g1[3] = (int)((tensor_h >> 16) | (tile_w_elems << 16));          // tile_dim0
  g1[4] = (int)(tile_h & 0xFFFFu);              // tile_dim1 (tile_dim2 = 0)
  g1[5] = (int)row_stride_elems;                // tensor_dim0_stride[31:0]
  g1[6] = 0;
  g1[7] = 0;
  v4i z4 = {0, 0, 0, 0};
#if __has_include(<hip/amd_detail/amd_gfx1250_TDM.h>)
  v8i z8 = {0, 0, 0, 0, 0, 0, 0, 0};
  __builtin_amdgcn_tensor_load_to_lds(g0, g1, z4, z4, z8, 0);
#else
  __builtin_amdgcn_tensor_load_to_lds(g0, g1, z4, z4, 0);
#endif
}

// ---------------------------------------------------------------------------
// One-time conversions
// ---------------------------------------------------------------------------
__global__ __launch_bounds__(256) void cvt_f32_bf16(
    const float* __restrict__ in, __bf16* __restrict__ out, int n)
{
  int i = (blockIdx.x * 256 + threadIdx.x) * 8;
  if (i + 8 <= n) {
    float4 a = *(const float4*)(in + i);
    float4 b = *(const float4*)(in + i + 4);
    v8bf o;
    o[0] = (__bf16)a.x; o[1] = (__bf16)a.y; o[2] = (__bf16)a.z; o[3] = (__bf16)a.w;
    o[4] = (__bf16)b.x; o[5] = (__bf16)b.y; o[6] = (__bf16)b.z; o[7] = (__bf16)b.w;
    *(v8bf*)(out + i) = o;
  }
}

// W[K][N] fp32 -> Wt[N][K] bf16, 32x32 tiles through LDS
__global__ __launch_bounds__(256) void transpose_cvt(
    const float* __restrict__ W, __bf16* __restrict__ Wt, int K, int N)
{
  __shared__ __bf16 t[32][33];
  const int n0 = blockIdx.x * 32, k0 = blockIdx.y * 32;
  const int tx = threadIdx.x & 31, ty = threadIdx.x >> 5;   // 32 x 8
#pragma unroll
  for (int i = 0; i < 32; i += 8)
    t[ty + i][tx] = (__bf16)W[(size_t)(k0 + ty + i) * N + n0 + tx];
  __syncthreads();
#pragma unroll
  for (int i = 0; i < 32; i += 8)
    Wt[(size_t)(n0 + ty + i) * K + k0 + tx] = t[tx][ty + i];
}

// ---------------------------------------------------------------------------
// GEMM: C[M,N] = A[M,K](bf16) @ Wt[N,K](bf16)^T + bias, TDM-staged, dbl-buf.
// BM=128, BN=128, BK=32; 8 waves (2x4); wave tile 64x32.
// ---------------------------------------------------------------------------
#define BM 128
#define BN 128
#define BK 32
#define LDK (BK + 8)   // 64B tile row + 16B TDM pad

__global__ __launch_bounds__(256) void gemm_bf16_tdm(
    const __bf16* __restrict__ A, const __bf16* __restrict__ Wt,
    const float* __restrict__ bias, void* __restrict__ Cout,
    int M, int N, int K, float out_scale, int out_is_bf16)
{
  __shared__ __bf16 As[2][BM][LDK];   // [m][k]
  __shared__ __bf16 Ws[2][BN][LDK];   // [n][k]

  const int tid  = threadIdx.x;
  const int lane = tid & 31;
  const int w    = tid >> 5;
  const int half = lane >> 4;
  const int lm   = lane & 15;
  const int wm   = (w >> 2) * 64;
  const int wn   = (w & 3)  * 32;
  const int m0   = blockIdx.y * BM;
  const int n0   = blockIdx.x * BN;

  v8f acc[4][2];
#pragma unroll
  for (int i = 0; i < 4; ++i)
#pragma unroll
    for (int j = 0; j < 2; ++j)
      acc[i][j] = (v8f){0.f, 0.f, 0.f, 0.f, 0.f, 0.f, 0.f, 0.f};

  auto issue = [&](int buf, int kk0) {
    tdm_load_tile_2d((u32)(uintptr_t)&As[buf][0][0],
                     A + (size_t)m0 * K + kk0,
                     BK, BM, (u32)K, (u32)K, (u32)M, /*pad*/3, 3);
    tdm_load_tile_2d((u32)(uintptr_t)&Ws[buf][0][0],
                     Wt + (size_t)n0 * K + kk0,
                     BK, BN, (u32)K, (u32)K, (u32)N, /*pad*/3, 3);
  };

  if (w == 0) {
    issue(0, 0);
    __builtin_amdgcn_s_wait_tensorcnt(0);
  }
  __syncthreads();

  const int nk = K / BK;
  for (int kt = 0; kt < nk; ++kt) {
    const int cur = kt & 1;
    if (kt + 1 < nk && w == 0) issue(cur ^ 1, (kt + 1) * BK);

    v16bf bf[2];
#pragma unroll
    for (int j = 0; j < 2; ++j)
      bf[j] = load_frag_b(&Ws[cur][wn + j * 16 + lm][0], half);
#pragma unroll
    for (int i = 0; i < 4; ++i) {
      v16bf af = load_frag_a(&As[cur][wm + i * 16 + lm][0], half);
#pragma unroll
      for (int j = 0; j < 2; ++j)
        acc[i][j] = wmma_bf16(af, bf[j], acc[i][j]);
    }

    if (w == 0) __builtin_amdgcn_s_wait_tensorcnt(0);
    __syncthreads();
  }

  // epilogue: C layout row = r + 8*half, col = lane&15
#pragma unroll
  for (int i = 0; i < 4; ++i) {
#pragma unroll
    for (int j = 0; j < 2; ++j) {
      const int col = n0 + wn + j * 16 + lm;
      const float bv = bias[col];
      if (out_is_bf16) {
        __bf16* C = (__bf16*)Cout;
#pragma unroll
        for (int r = 0; r < 8; ++r) {
          const int row = m0 + wm + i * 16 + r + 8 * half;
          C[(size_t)row * N + col] = (__bf16)((acc[i][j][r] + bv) * out_scale);
        }
      } else {
        float* C = (float*)Cout;
#pragma unroll
        for (int r = 0; r < 8; ++r) {
          const int row = m0 + wm + i * 16 + r + 8 * half;
          C[(size_t)row * N + col] = (acc[i][j][r] + bv) * out_scale;
        }
      }
    }
  }
}

// ---------------------------------------------------------------------------
// Flash attention (GQA), all-bf16 operands. One block per (b, head, 64 q's).
// 4 waves x 16 query rows. K tile TDM-loaded; V staged transposed by threads.
// 1/sqrt(d) pre-folded into qh by the Q-projection epilogue.
// ---------------------------------------------------------------------------
__global__ __launch_bounds__(128) void gqa_flash_wmma(
    const __bf16* __restrict__ qh, const __bf16* __restrict__ kh,
    const __bf16* __restrict__ vh, __bf16* __restrict__ ao)
{
  __shared__ __bf16 Ks[64][72];        // [key][d]   (TDM, 128B row + 16B pad)
  __shared__ __bf16 Vt[64][72];        // [d][key]
  __shared__ __bf16 Ps[4][16][72];     // per-wave P tile [m][key]

  const int tid  = threadIdx.x;
  const int lane = tid & 31;
  const int w    = tid >> 5;
  const int half = lane >> 4;
  const int lm   = lane & 15;

  const int head = blockIdx.y;
  const int g    = head >> 2;
  const int b    = blockIdx.z;
  const int q0   = blockIdx.x * 64;

  // persistent Q fragments straight from global bf16
  const __bf16* qrow = qh + ((size_t)b * SEQ + q0 + w * 16 + lm) * D_MODEL
                          + head * HEAD_DIM;
  v16bf qf[2];
#pragma unroll
  for (int f = 0; f < 2; ++f)
    qf[f] = frag_pair(qrow + f * 32 + 8 * half, qrow + f * 32 + 16 + 8 * half);

  float mrow[8], lrow[8];
  v8f oacc[4];
#pragma unroll
  for (int r = 0; r < 8; ++r) { mrow[r] = -3.0e38f; lrow[r] = 0.f; }
#pragma unroll
  for (int d = 0; d < 4; ++d)
    oacc[d] = (v8f){0.f, 0.f, 0.f, 0.f, 0.f, 0.f, 0.f, 0.f};

  const __bf16* kbase = kh + (size_t)b * SEQ * KV_DIM + g * HEAD_DIM;
  const __bf16* vbase = vh + (size_t)b * SEQ * KV_DIM + g * HEAD_DIM;

  for (int t0 = 0; t0 < SEQ; t0 += 64) {
    // async K tile while threads stage V transposed
    if (w == 0)
      tdm_load_tile_2d((u32)(uintptr_t)&Ks[0][0],
                       kbase + (size_t)t0 * KV_DIM,
                       HEAD_DIM, 64, KV_DIM, KV_DIM, (u32)SEQ, /*pad*/4, 3);
    {
      const int key = tid >> 1;
      const int dc  = (tid & 1) * 32;
      const __bf16* sv = vbase + (size_t)(t0 + key) * KV_DIM + dc;
#pragma unroll
      for (int i = 0; i < 32; i += 8) {
        v8bf x = *(const v8bf*)(sv + i);
#pragma unroll
        for (int j = 0; j < 8; ++j) Vt[dc + i + j][key] = x[j];
      }
    }
    if (w == 0) __builtin_amdgcn_s_wait_tensorcnt(0);
    __syncthreads();

    // S = Q @ K^T  (Q pre-scaled)
    v8f sacc[4];
#pragma unroll
    for (int a = 0; a < 4; ++a) {
      v8f s = (v8f){0.f, 0.f, 0.f, 0.f, 0.f, 0.f, 0.f, 0.f};
#pragma unroll
      for (int f = 0; f < 2; ++f) {
        v16bf bfr = load_frag_b(&Ks[a * 16 + lm][f * 32], half);
        s = wmma_bf16(qf[f], bfr, s);
      }
      sacc[a] = s;
    }

    // online softmax (row r in lanes of the same 16-lane half)
#pragma unroll
    for (int r = 0; r < 8; ++r) {
      float mx = fmaxf(fmaxf(sacc[0][r], sacc[1][r]),
                       fmaxf(sacc[2][r], sacc[3][r]));
      mx = fmaxf(mx, __shfl_xor(mx, 1));
      mx = fmaxf(mx, __shfl_xor(mx, 2));
      mx = fmaxf(mx, __shfl_xor(mx, 4));
      mx = fmaxf(mx, __shfl_xor(mx, 8));
      const float mnew  = fmaxf(mrow[r], mx);
      const float alpha = __expf(mrow[r] - mnew);
      mrow[r] = mnew;
      float psum = 0.f;
#pragma unroll
      for (int a = 0; a < 4; ++a) {
        float p = __expf(sacc[a][r] - mnew);
        psum += p;
        Ps[w][r + 8 * half][a * 16 + lm] = (__bf16)p;
      }
      psum += __shfl_xor(psum, 1);
      psum += __shfl_xor(psum, 2);
      psum += __shfl_xor(psum, 4);
      psum += __shfl_xor(psum, 8);
      lrow[r] = lrow[r] * alpha + psum;
#pragma unroll
      for (int d = 0; d < 4; ++d) oacc[d][r] *= alpha;
    }

    asm volatile("s_wait_dscnt 0" ::: "memory");   // Ps stores -> loads, same wave

    // O += P @ V
    v16bf pf[2];
#pragma unroll
    for (int f = 0; f < 2; ++f)
      pf[f] = load_frag_a(&Ps[w][lm][f * 32], half);
#pragma unroll
    for (int d = 0; d < 4; ++d) {
#pragma unroll
      for (int f = 0; f < 2; ++f) {
        v16bf vf = load_frag_b(&Vt[d * 16 + lm][f * 32], half);
        oacc[d] = wmma_bf16(pf[f], vf, oacc[d]);
      }
    }
    __syncthreads();   // done with Ks/Vt before next tile overwrites
  }

  // normalize + store bf16
  __bf16* obase = ao + ((size_t)b * SEQ + q0 + w * 16) * D_MODEL
                     + head * HEAD_DIM;
#pragma unroll
  for (int r = 0; r < 8; ++r) {
    const float inv = 1.0f / lrow[r];
    const int row = r + 8 * half;
#pragma unroll
    for (int d = 0; d < 4; ++d)
      obase[(size_t)row * D_MODEL + d * 16 + lm] = (__bf16)(oacc[d][r] * inv);
  }
}

// ---------------------------------------------------------------------------
extern "C" void kernel_launch(void* const* d_in, const int* in_sizes, int n_in,
                              void* d_out, int out_size, void* d_ws, size_t ws_size,
                              hipStream_t stream) {
  const float* q  = (const float*)d_in[0];
  const float* k  = (const float*)d_in[1];
  const float* v  = (const float*)d_in[2];
  const float* Wq = (const float*)d_in[3];
  const float* bq = (const float*)d_in[4];
  const float* Wk = (const float*)d_in[5];
  const float* bk = (const float*)d_in[6];
  const float* Wv = (const float*)d_in[7];
  const float* bv = (const float*)d_in[8];
  const float* Wo = (const float*)d_in[9];
  const float* bo = (const float*)d_in[10];

  __bf16* qb  = (__bf16*)d_ws;                        // activations, bf16
  __bf16* kb  = qb  + (size_t)ROWS * D_MODEL;
  __bf16* vb  = kb  + (size_t)ROWS * D_MODEL;
  __bf16* Wqt = vb  + (size_t)ROWS * D_MODEL;         // weights, transposed bf16
  __bf16* Wkt = Wqt + (size_t)D_MODEL * D_MODEL;
  __bf16* Wvt = Wkt + (size_t)KV_DIM * D_MODEL;
  __bf16* Wot = Wvt + (size_t)KV_DIM * D_MODEL;
  __bf16* qhp = Wot + (size_t)D_MODEL * D_MODEL;      // projections, bf16
  __bf16* khp = qhp + (size_t)ROWS * D_MODEL;
  __bf16* vhp = khp + (size_t)ROWS * KV_DIM;
  __bf16* aop = vhp + (size_t)ROWS * KV_DIM;          // attention out, bf16

  const int nact = ROWS * D_MODEL;
  cvt_f32_bf16<<<nact / (256 * 8), 256, 0, stream>>>(q, qb, nact);
  cvt_f32_bf16<<<nact / (256 * 8), 256, 0, stream>>>(k, kb, nact);
  cvt_f32_bf16<<<nact / (256 * 8), 256, 0, stream>>>(v, vb, nact);
  transpose_cvt<<<dim3(D_MODEL / 32, D_MODEL / 32), 256, 0, stream>>>(Wq, Wqt, D_MODEL, D_MODEL);
  transpose_cvt<<<dim3(KV_DIM  / 32, D_MODEL / 32), 256, 0, stream>>>(Wk, Wkt, D_MODEL, KV_DIM);
  transpose_cvt<<<dim3(KV_DIM  / 32, D_MODEL / 32), 256, 0, stream>>>(Wv, Wvt, D_MODEL, KV_DIM);
  transpose_cvt<<<dim3(D_MODEL / 32, D_MODEL / 32), 256, 0, stream>>>(Wo, Wot, D_MODEL, D_MODEL);

  // projections (1/sqrt(HEAD_DIM) folded into Q projection)
  gemm_bf16_tdm<<<dim3(D_MODEL / BN, ROWS / BM), 256, 0, stream>>>(
      qb, Wqt, bq, qhp, ROWS, D_MODEL, D_MODEL, 0.125f, 1);
  gemm_bf16_tdm<<<dim3(KV_DIM / BN, ROWS / BM), 256, 0, stream>>>(
      kb, Wkt, bk, khp, ROWS, KV_DIM, D_MODEL, 1.0f, 1);
  gemm_bf16_tdm<<<dim3(KV_DIM / BN, ROWS / BM), 256, 0, stream>>>(
      vb, Wvt, bv, vhp, ROWS, KV_DIM, D_MODEL, 1.0f, 1);

  gqa_flash_wmma<<<dim3(SEQ / 64, NUM_HEADS, BATCH), dim3(128), 0, stream>>>(
      qhp, khp, vhp, aop);

  gemm_bf16_tdm<<<dim3(D_MODEL / BN, ROWS / BM), 256, 0, stream>>>(
      aop, Wot, bo, d_out, ROWS, D_MODEL, D_MODEL, 1.0f, 0);
}
